// GA_GAT_40724879900765
// MI455X (gfx1250) — matrix operand order repaired and checked by reading, not compiled, not run on hardware
//
#include <hip/hip_runtime.h>
#include <stdint.h>

#define HID 64
#define HEADS 4
#define HD 16
#define LAYERS 4
#define NGRAPHS 256
#define NUM_RBF 20
#define CUTOFF_F 10.0f
#define NEG_SLOPE 0.2f
#define EPS_LN 1e-6f

typedef __attribute__((ext_vector_type(16))) __bf16 v16bf;
typedef __attribute__((ext_vector_type(8)))  float  v8f;
typedef __attribute__((ext_vector_type(8)))  unsigned v8u;
typedef __attribute__((ext_vector_type(4)))  unsigned v4u;

// ---------- helpers ----------
__device__ __forceinline__ int gradeOf(int b) { return (b == 0) ? 0 : (b < 4) ? 1 : (b < 7) ? 2 : 3; }

__device__ __forceinline__ unsigned short f2bfu(float f) {
    unsigned u = __builtin_bit_cast(unsigned, f);
    return (unsigned short)(u >> 16);
}
__device__ __forceinline__ float bf2f(unsigned short h) {
    unsigned u = ((unsigned)h) << 16;
    return __builtin_bit_cast(float, u);
}
// pack two f32 -> one dword of two bf16 (lo in [15:0], hi in [31:16])
__device__ __forceinline__ unsigned pk2bf(float lo, float hi) {
    unsigned a = __builtin_bit_cast(unsigned, lo);
    unsigned b = __builtin_bit_cast(unsigned, hi);
    return (a >> 16) | (b & 0xffff0000u);
}
// order-preserving float<->uint mapping for atomicMax on floats
__device__ __forceinline__ unsigned f2ord(float f) {
    unsigned u = __builtin_bit_cast(unsigned, f);
    return (u & 0x80000000u) ? ~u : (u | 0x80000000u);
}
__device__ __forceinline__ float ord2f(unsigned o) {
    unsigned u = (o & 0x80000000u) ? (o & 0x7fffffffu) : ~o;
    return __builtin_bit_cast(float, u);
}
__device__ __forceinline__ float sigmoidf_(float x) { return 1.0f / (1.0f + __expf(-x)); }
__device__ __forceinline__ float siluf_(float x) { return x * sigmoidf_(x); }

// ---------- utility kernels ----------
__global__ void k_zero_f32(float* p, long n) {
    long t = (long)blockIdx.x * blockDim.x + threadIdx.x;
    if (t < n) p[t] = 0.0f;
}
__global__ void k_zero_f4(float4* p, long n4) {
    long t = (long)blockIdx.x * blockDim.x + threadIdx.x;
    if (t < n4) p[t] = make_float4(0.f, 0.f, 0.f, 0.f);
}
__global__ void k_init_edge_stats(unsigned* emax, float* asum, long n) {
    long t = (long)blockIdx.x * blockDim.x + threadIdx.x;
    if (t < n) { emax[t] = f2ord(-3.0e38f); asum[t] = 0.0f; }
}

// ---------- graph statistics (centering) ----------
__global__ void k_graph_stats(const float* __restrict__ pos, const int* __restrict__ batch,
                              float* cnt, float* possum, int N) {
    int n = blockIdx.x * blockDim.x + threadIdx.x;
    if (n >= N) return;
    int b = batch[n];
    atomicAdd(&cnt[b], 1.0f);
    atomicAdd(&possum[b * 3 + 0], pos[n * 3 + 0]);
    atomicAdd(&possum[b * 3 + 1], pos[n * 3 + 1]);
    atomicAdd(&possum[b * 3 + 2], pos[n * 3 + 2]);
}
__global__ void k_pos_mean(const float* cnt, const float* possum, float* pmean) {
    int b = threadIdx.x;  // 256 threads
    float c = fmaxf(cnt[b], 1.0f);
    pmean[b * 3 + 0] = possum[b * 3 + 0] / c;
    pmean[b * 3 + 1] = possum[b * 3 + 1] / c;
    pmean[b * 3 + 2] = possum[b * 3 + 2] / c;
}

// ---------- node embedding + input projection ----------
__global__ void k_embed(const float* __restrict__ pos, const int* __restrict__ zat,
                        const int* __restrict__ batch, const float* __restrict__ emb,
                        const float* __restrict__ inw, const float* __restrict__ inb,
                        const float* __restrict__ pmean, float* __restrict__ h, int N) {
    long t = (long)blockIdx.x * blockDim.x + threadIdx.x;
    if (t >= (long)N * HID) return;
    int n = (int)(t / HID), o = (int)(t % HID);
    const float* er = emb + (long)zat[n] * HID;
    const float* wr = inw + (long)o * (HID + 1);
    float acc = inb[o];
#pragma unroll 8
    for (int i = 0; i < HID; i++) acc += er[i] * wr[i];
    float wv = wr[HID];
    int bb = batch[n];
    float px = pos[n * 3 + 0] - pmean[bb * 3 + 0];
    float py = pos[n * 3 + 1] - pmean[bb * 3 + 1];
    float pz = pos[n * 3 + 2] - pmean[bb * 3 + 2];
    float* hp = h + ((long)n * HID + o) * 8;
    hp[0] = acc; hp[1] = wv * px; hp[2] = wv * py; hp[3] = wv * pz;
    hp[4] = 0.f; hp[5] = 0.f; hp[6] = 0.f; hp[7] = 0.f;
}

// ---------- per-edge RBF MLP -> per-head scalar bias (loop-invariant over layers) ----------
__global__ void k_edge_mlp(const float* __restrict__ pos, const int* __restrict__ ei,
                           const float* __restrict__ ew1, const float* __restrict__ eb1,
                           const float* __restrict__ ew2, const float* __restrict__ eb2,
                           float* __restrict__ escal, int E) {
    int e = blockIdx.x * blockDim.x + threadIdx.x;
    if (e >= E) return;
    int s = ei[e], d = ei[E + e];
    float dx = pos[s * 3 + 0] - pos[d * 3 + 0];
    float dy = pos[s * 3 + 1] - pos[d * 3 + 1];
    float dz = pos[s * 3 + 2] - pos[d * 3 + 2];
    float d2 = dx * dx + dy * dy + dz * dz;
    float dist = (d2 > 0.f) ? sqrtf(d2) : 0.f;
    const float width = CUTOFF_F / (NUM_RBF - 1);
    float rbf[NUM_RBF];
#pragma unroll
    for (int j = 0; j < NUM_RBF; j++) {
        float u = (dist - (float)j * width) / width;
        rbf[j] = __expf(-0.5f * u * u);
    }
    float hs[HEADS] = {eb2[0], eb2[1], eb2[2], eb2[3]};
    for (int o = 0; o < HID; o++) {
        float a = eb1[o];
        const float* w1 = ew1 + o * NUM_RBF;
#pragma unroll
        for (int j = 0; j < NUM_RBF; j++) a += rbf[j] * w1[j];
        a = siluf_(a);
#pragma unroll
        for (int hh = 0; hh < HEADS; hh++) hs[hh] += a * ew2[hh * HID + o];
    }
#pragma unroll
    for (int hh = 0; hh < HEADS; hh++) escal[(long)e * 4 + hh] = hs[hh];
}

// ---------- MVLinear via WMMA + TDM tile staging ----------
// One wave per (16-node tile, blade). Input tile (16 nodes x 512 f32 = 32KB contiguous)
// is DMA'd into LDS by the Tensor Data Mover; outputs are repacked through LDS for
// coalesced bf16 stores.
__global__ void __launch_bounds__(256) k_mvlinear_wmma(
        const float* __restrict__ h, const float* __restrict__ pw, const float* __restrict__ pb,
        unsigned short* __restrict__ zout, int N) {
    __shared__ __align__(16) float tile[16 * HID * 8];           // 32 KB input stage
    __shared__ __align__(16) unsigned short otile[16 * HID * 8]; // 16 KB output stage
    const int tid = threadIdx.x;
    const int tb = blockIdx.x * 16;

    // --- TDM: DMA the 2D tile (rows = nodes, 512 f32 each) global -> LDS.
    // tensor_dim1 = remaining rows, so the tail tile is zero-filled by TDM OOB rule.
    if (tid < 32) {  // one wave issues; EXEC is ignored by TENSOR ops
        unsigned lds_off = (unsigned)(uintptr_t)(&tile[0]);
        unsigned long long ga = (unsigned long long)(uintptr_t)(h + (size_t)tb * 512);
        unsigned nrem = (unsigned)(N - tb);
        v4u g0;
        g0[0] = 1u;                                              // count=1, user desc
        g0[1] = lds_off;                                         // lds_addr
        g0[2] = (unsigned)(ga & 0xffffffffu);                    // global_addr[31:0]
        g0[3] = (unsigned)((ga >> 32) & 0x01ffffffu) | (2u << 30); // addr[56:32] | type=2
        v8u g1;
        g1[0] = (2u << 16);                  // workgroup_mask=0, data_size=2 (4B)
        g1[1] = (512u << 16);                // tensor_dim0[15:0]=512 -> bits[63:48]
        g1[2] = (nrem & 0xffffu) << 16;      // tensor_dim0 hi=0 | tensor_dim1 lo16
        g1[3] = (nrem >> 16) | (512u << 16); // tensor_dim1 hi16 | tile_dim0=512
        g1[4] = 16u;                         // tile_dim1=16, tile_dim2=0
        g1[5] = 512u;                        // tensor_dim0_stride lo32 = 512 elems
        g1[6] = 0u;                          // stride hi | dim1_stride lo (unused)
        g1[7] = 0u;
        asm volatile("tensor_load_to_lds %0, %1" :: "s"(g0), "s"(g1) : "memory");
        __builtin_amdgcn_s_wait_tensorcnt(0);
    }
    __syncthreads();

    const int lane  = tid & 31;
    const int blade = tid >> 5;            // 8 waves = 8 blades
    const int g     = gradeOf(blade);
    const int hi    = (lane >= 16) ? 1 : 0;
    const int ln15  = lane & 15;

    // A fragments (16x32 bf16, ISA layout). Dword d holds K0=(d<4 ? 2d : 2d+8)+hi*8, K0+1.
    v8u ua0, ua1;
#pragma unroll
    for (int d = 0; d < 8; d++) {
        int K0 = ((d < 4) ? 2 * d : 2 * d + 8) + hi * 8;
        const float* row = tile + (ln15 * HID) * 8 + blade;
        ua0[d] = pk2bf(row[K0 * 8], row[(K0 + 1) * 8]);
        ua1[d] = pk2bf(row[(K0 + 32) * 8], row[(K0 + 33) * 8]);
    }
    v16bf a0 = __builtin_bit_cast(v16bf, ua0);
    v16bf a1 = __builtin_bit_cast(v16bf, ua1);

    const float* W = pw + (long)g * HID * HID;  // W[o,i]; B[k,o] = W[o,k]

#pragma unroll
    for (int ot = 0; ot < 4; ot++) {
        int o = ot * 16 + ln15;
        // B fragments (32x16): dword d holds K0 = (d<4 ? 2d : 2d+8) + hi*8, K0+1 (adjacent in memory).
        v8u ub0, ub1;
        const float* wr = W + o * HID;
#pragma unroll
        for (int d = 0; d < 8; d++) {
            int K0 = ((d < 4) ? 2 * d : 2 * d + 8) + hi * 8;
            float2 p0 = *(const float2*)(wr + K0);
            float2 p1 = *(const float2*)(wr + K0 + 32);
            ub0[d] = pk2bf(p0.x, p0.y);
            ub1[d] = pk2bf(p1.x, p1.y);
        }
        v16bf b0 = __builtin_bit_cast(v16bf, ub0);
        v16bf b1 = __builtin_bit_cast(v16bf, ub1);

        float bias = (blade == 0) ? pb[o] : 0.0f;
        v8f c;
#pragma unroll
        for (int r = 0; r < 8; r++) c[r] = bias;
        c = __builtin_amdgcn_wmma_f32_16x16x32_bf16(false, a0, false, b0, (short)0, c, false, false);
        c = __builtin_amdgcn_wmma_f32_16x16x32_bf16(false, a1, false, b1, (short)0, c, false, false);

        // D layout: lane%16 = col(o), vgpr r (+8 for hi half) = row(node). Stage to LDS.
#pragma unroll
        for (int r = 0; r < 8; r++) {
            int nl = r + hi * 8;
            otile[(nl * HID + o) * 8 + blade] = f2bfu(c[r]);
        }
    }
    __syncthreads();

    // coalesced store: each thread moves 32 contiguous bf16 (64B) of the 16KB tile
    {
        int node = (tid * 32) >> 9;  // /512
        if (tb + node < N) {
            const uint4* src = (const uint4*)(otile + tid * 32);
            uint4* dst = (uint4*)(zout + (size_t)tb * 512 + tid * 32);
            dst[0] = src[0]; dst[1] = src[1]; dst[2] = src[2]; dst[3] = src[3];
        }
    }
}

// ---------- per-node attention pre-scores ----------
__global__ void k_scores(const unsigned short* __restrict__ zb,
                         const float* __restrict__ a_src, const float* __restrict__ a_dst,
                         const float* __restrict__ w_src, const float* __restrict__ w_dst,
                         float* __restrict__ ssrc, float* __restrict__ sdst, int N) {
    int n = blockIdx.x * blockDim.x + threadIdx.x;
    if (n >= N) return;
    for (int hh = 0; hh < HEADS; hh++) {
        float ss = 0.f, sd = 0.f;
        for (int c = 0; c < HD; c++) {
            long base = ((long)n * HID + hh * HD + c) * 8;
#pragma unroll
            for (int b = 0; b < 8; b++) {
                float v = bf2f(zb[base + b]);
                int gg = gradeOf(b);
                ss += v * a_src[((hh * HD) + c) * 8 + b] * w_src[hh * 4 + gg];
                sd += v * a_dst[((hh * HD) + c) * 8 + b] * w_dst[hh * 4 + gg];
            }
        }
        ssrc[n * 4 + hh] = ss;
        sdst[n * 4 + hh] = sd;
    }
}

// ---------- edge softmax pass 1: leaky-relu scores + segment max ----------
__global__ void k_edge_max(const int* __restrict__ ei, const float* __restrict__ ssrc,
                           const float* __restrict__ sdst, const float* __restrict__ escal,
                           float* __restrict__ ealpha, unsigned* __restrict__ emax, int E) {
    long t = (long)blockIdx.x * blockDim.x + threadIdx.x;
    if (t >= (long)E * 4) return;
    int e = (int)(t >> 2), hh = (int)(t & 3);
    int s = ei[e], d = ei[E + e];
    float v = ssrc[s * 4 + hh] + sdst[d * 4 + hh] + escal[(long)e * 4 + hh];
    v = (v > 0.f) ? v : NEG_SLOPE * v;
    ealpha[t] = v;
    atomicMax(&emax[d * 4 + hh], f2ord(v));
}

// ---------- edge softmax pass 2: exp + segment sum ----------
__global__ void k_edge_sum(const int* __restrict__ ei, const unsigned* __restrict__ emax,
                           float* __restrict__ ealpha, float* __restrict__ asum, int E) {
    long t = (long)blockIdx.x * blockDim.x + threadIdx.x;
    if (t >= (long)E * 4) return;
    int e = (int)(t >> 2), hh = (int)(t & 3);
    int d = ei[E + e];
    float a = __expf(ealpha[t] - ord2f(emax[d * 4 + hh]));
    ealpha[t] = a;
    atomicAdd(&asum[d * 4 + hh], a);
}

// ---------- message scatter: one wave per edge, lane handles 16 contiguous comps ----------
__global__ void __launch_bounds__(256) k_scatter(
        const int* __restrict__ ei, const unsigned short* __restrict__ zb,
        const float* __restrict__ ealpha, const float* __restrict__ asum,
        float* __restrict__ attout, int E) {
    int e = blockIdx.x * 8 + (threadIdx.x >> 5);
    if (e >= E) return;
    int lane = threadIdx.x & 31;
    int s = ei[e], d = ei[E + e];
    int hh = lane >> 3;  // 16 comps = 2 channels of one head
    float alpha = ealpha[(long)e * 4 + hh] / (asum[d * 4 + hh] + 1e-16f);
    const unsigned short* zp = zb + (long)s * 512 + lane * 16;
    float* op = attout + (long)d * 512 + lane * 16;
    uint4 p0 = *(const uint4*)(zp);
    uint4 p1 = *(const uint4*)(zp + 8);
    unsigned w[8] = {p0.x, p0.y, p0.z, p0.w, p1.x, p1.y, p1.z, p1.w};
#pragma unroll
    for (int q = 0; q < 8; q++) {
        atomicAdd(op + 2 * q,     bf2f((unsigned short)(w[q] & 0xffffu)) * alpha);
        atomicAdd(op + 2 * q + 1, bf2f((unsigned short)(w[q] >> 16)) * alpha);
    }
}

// ---------- fused mv-SiLU + residual + mv-LayerNorm: one wave per node ----------
__global__ void __launch_bounds__(256) k_node_update(
        const float* __restrict__ attout, float* __restrict__ h,
        const float* __restrict__ sa, const float* __restrict__ sb,
        const float* __restrict__ lna, int N) {
    int n = blockIdx.x * 8 + (threadIdx.x >> 5);
    int lane = threadIdx.x & 31;
    bool valid = (n < N);
    float y[2][8];
    float nsum = 0.0f;
#pragma unroll
    for (int half = 0; half < 2; half++) {
        int c = lane + half * 32;
        if (valid) {
            const float* xp = attout + ((long)n * HID + c) * 8;
            const float* hp = h + ((long)n * HID + c) * 8;
            float x[8];
#pragma unroll
            for (int b = 0; b < 8; b++) x[b] = xp[b];
            float norms[4];
            norms[0] = x[0];
            norms[1] = x[1] * x[1] + x[2] * x[2] + x[3] * x[3];
            norms[2] = x[4] * x[4] + x[5] * x[5] + x[6] * x[6];
            norms[3] = x[7] * x[7];
            float gate[4];
#pragma unroll
            for (int gg = 0; gg < 4; gg++) gate[gg] = sigmoidf_(sa[c * 4 + gg] * norms[gg] + sb[c * 4 + gg]);
            float ss = 0.f;
#pragma unroll
            for (int b = 0; b < 8; b++) {
                float v = gate[gradeOf(b)] * x[b] + hp[b];
                y[half][b] = v;
                ss += v * v;
            }
            nsum += sqrtf(ss);
        }
    }
#pragma unroll
    for (int m = 16; m > 0; m >>= 1) nsum += __shfl_xor(nsum, m, 32);  // wave32 reduction
    float denom = nsum / (float)HID + EPS_LN;
    if (valid) {
#pragma unroll
        for (int half = 0; half < 2; half++) {
            int c = lane + half * 32;
            float scale = lna[c] / denom;
            float* hp = h + ((long)n * HID + c) * 8;
#pragma unroll
            for (int b = 0; b < 8; b++) hp[b] = scale * y[half][b];
        }
    }
}

// ---------- prepool scalar-blade projection + graph readout ----------
__global__ void k_prepool(const float* __restrict__ h, const int* __restrict__ batch,
                          const float* __restrict__ ppw, const float* __restrict__ ppb,
                          float* __restrict__ gbuf, int N) {
    long t = (long)blockIdx.x * blockDim.x + threadIdx.x;
    if (t >= (long)N * HID) return;
    int n = (int)(t / HID), o = (int)(t % HID);
    const float* W = ppw + (long)o * HID;  // grade-0 block (blade 0)
    float acc = ppb[o];
#pragma unroll 8
    for (int i = 0; i < HID; i++) acc += h[((long)n * HID + i) * 8] * W[i];
    atomicAdd(&gbuf[(long)batch[n] * HID + o], acc);
}

__global__ void k_final(const float* __restrict__ gbuf, const float* __restrict__ pw1,
                        const float* __restrict__ pb1, const float* __restrict__ pw2,
                        const float* __restrict__ pb2, float* __restrict__ out) {
    __shared__ float hid[HID];
    int b = blockIdx.x, o = threadIdx.x;
    float acc = pb1[o];
#pragma unroll 8
    for (int i = 0; i < HID; i++) acc += gbuf[(long)b * HID + i] * pw1[o * HID + i];
    hid[o] = siluf_(acc);
    __syncthreads();
    if (o == 0) {
        float s = pb2[0];
#pragma unroll 8
        for (int i = 0; i < HID; i++) s += hid[i] * pw2[i];
        out[b] = s;
    }
}

// ---------- host orchestration ----------
extern "C" void kernel_launch(void* const* d_in, const int* in_sizes, int n_in,
                              void* d_out, int out_size, void* d_ws, size_t ws_size,
                              hipStream_t stream) {
    (void)n_in; (void)out_size; (void)ws_size;
    const float* pos    = (const float*)d_in[0];
    const int*   zat    = (const int*)d_in[1];
    const int*   ei     = (const int*)d_in[2];
    const int*   batch  = (const int*)d_in[3];
    const float* emb    = (const float*)d_in[4];
    const float* inw    = (const float*)d_in[5];
    const float* inb    = (const float*)d_in[6];
    const float* ew1    = (const float*)d_in[7];
    const float* eb1    = (const float*)d_in[8];
    const float* ew2    = (const float*)d_in[9];
    const float* eb2    = (const float*)d_in[10];
    const float* proj_w = (const float*)d_in[11];
    const float* proj_b = (const float*)d_in[12];
    const float* a_src  = (const float*)d_in[13];
    const float* a_dst  = (const float*)d_in[14];
    const float* w_src  = (const float*)d_in[15];
    const float* w_dst  = (const float*)d_in[16];
    const float* ln_a   = (const float*)d_in[17];
    const float* silu_a = (const float*)d_in[18];
    const float* silu_b = (const float*)d_in[19];
    const float* ppw    = (const float*)d_in[20];
    const float* ppb    = (const float*)d_in[21];
    const float* pw1    = (const float*)d_in[22];
    const float* pb1    = (const float*)d_in[23];
    const float* pw2    = (const float*)d_in[24];
    const float* pb2    = (const float*)d_in[25];
    float* out = (float*)d_out;

    const int N = in_sizes[0] / 3;
    const int E = in_sizes[2] / 2;

    char* ws = (char*)d_ws;
    size_t off = 0;
    auto take = [&](size_t bytes) -> char* {
        char* p = ws + off;
        off = (off + bytes + 255) & ~(size_t)255;
        return p;
    };
    float*          h      = (float*)take((size_t)N * 512 * 4);
    float*          attout = (float*)take((size_t)N * 512 * 4);
    unsigned short* zb     = (unsigned short*)take((size_t)N * 512 * 2);
    float*          ssrc   = (float*)take((size_t)N * 4 * 4);
    float*          sdst   = (float*)take((size_t)N * 4 * 4);
    float*          ealpha = (float*)take((size_t)E * 4 * 4);
    float*          escal  = (float*)take((size_t)E * 4 * 4);
    unsigned*       emax   = (unsigned*)take((size_t)N * 4 * 4);
    float*          asum   = (float*)take((size_t)N * 4 * 4);
    float*          cnt    = (float*)take(256 * 4);
    float*          possum = (float*)take(768 * 4);
    float*          pmean  = (float*)take(768 * 4);
    float*          gbuf   = (float*)take((size_t)NGRAPHS * HID * 4);

    auto cdiv = [](long a, long b) { return (int)((a + b - 1) / b); };

    // graph centering + readout init
    k_zero_f32<<<cdiv(256, 256), 256, 0, stream>>>(cnt, 256);
    k_zero_f32<<<cdiv(768, 256), 256, 0, stream>>>(possum, 768);
    k_zero_f32<<<cdiv((long)NGRAPHS * HID, 256), 256, 0, stream>>>(gbuf, (long)NGRAPHS * HID);
    k_graph_stats<<<cdiv(N, 256), 256, 0, stream>>>(pos, batch, cnt, possum, N);
    k_pos_mean<<<1, 256, 0, stream>>>(cnt, possum, pmean);

    // node embedding + input projection
    k_embed<<<cdiv((long)N * HID, 256), 256, 0, stream>>>(pos, zat, batch, emb, inw, inb, pmean, h, N);

    // per-edge scalar bias (loop-invariant)
    k_edge_mlp<<<cdiv(E, 256), 256, 0, stream>>>(pos, ei, ew1, eb1, ew2, eb2, escal, E);

    for (int l = 0; l < LAYERS; l++) {
        const float* pwL  = proj_w + (size_t)l * 4 * HID * HID;
        const float* pbL  = proj_b + (size_t)l * HID;
        const float* asL  = a_src + (size_t)l * HEADS * HD * 8;
        const float* adL  = a_dst + (size_t)l * HEADS * HD * 8;
        const float* wsL  = w_src + (size_t)l * HEADS * 4;
        const float* wdL  = w_dst + (size_t)l * HEADS * 4;
        const float* lnL  = ln_a + (size_t)l * HID;
        const float* saL  = silu_a + (size_t)l * HID * 4;
        const float* sbL  = silu_b + (size_t)l * HID * 4;

        // TDM-staged WMMA per-grade projection: h (f32) -> zb (bf16 messages)
        k_mvlinear_wmma<<<cdiv(N, 16), 256, 0, stream>>>(h, pwL, pbL, zb, N);

        // attention pre-scores
        k_scores<<<cdiv(N, 256), 256, 0, stream>>>(zb, asL, adL, wsL, wdL, ssrc, sdst, N);

        // reset per-layer accumulators
        k_zero_f4<<<cdiv((long)N * 128, 256), 256, 0, stream>>>((float4*)attout, (long)N * 128);
        k_init_edge_stats<<<cdiv((long)N * 4, 256), 256, 0, stream>>>(emax, asum, (long)N * 4);

        // edge softmax (segment max, then exp + segment sum)
        k_edge_max<<<cdiv((long)E * 4, 256), 256, 0, stream>>>(ei, ssrc, sdst, escal, ealpha, emax, E);
        k_edge_sum<<<cdiv((long)E * 4, 256), 256, 0, stream>>>(ei, emax, ealpha, asum, E);

        // weighted message scatter (one wave per edge)
        k_scatter<<<cdiv((long)E, 8), 256, 0, stream>>>(ei, zb, ealpha, asum, attout, E);

        // mv-SiLU + residual + mv-LayerNorm (one wave per node), writes h in place
        k_node_update<<<cdiv((long)N, 8), 256, 0, stream>>>(attout, h, saL, sbL, lnL, N);
    }

    // scalar-blade prepool + graph sum + final MLP
    k_prepool<<<cdiv((long)N * HID, 256), 256, 0, stream>>>(h, batch, ppw, ppb, gbuf, N);
    k_final<<<NGRAPHS, HID, 0, stream>>>(gbuf, pw1, pb1, pw2, pb2, out);
}
